// CRF_86517821212164
// MI455X (gfx1250) — matrix operand order repaired and checked by reading, not compiled, not run on hardware
//
#include <hip/hip_runtime.h>
#include <hip/hip_bf16.h>

typedef __attribute__((ext_vector_type(16))) _Float16 v16h;
typedef __attribute__((ext_vector_type(8)))  float    v8f;

#define BB  512
#define TT  1024
#define CC  64
#define CP2 66          // C+2 (transition matrix row stride)
#define START_TAG 64
#define END_TAG   65

// ---------------------------------------------------------------------------
// Forward algorithm: one wave handles 16 sequences (M dim of WMMA).
// Per step t:  new[k] = m_b + x_t[b][k] + log( sum_j exp(fwd[b][j]-m_b)*ET[j][k] )
// The GEMV over j is done with v_wmma_f32_16x16x32_f16 (8 WMMA / step).
// ---------------------------------------------------------------------------
__global__ __launch_bounds__(32) void crf_forward_kernel(
    const float* __restrict__ inputs,   // (B,T,C)
    const float* __restrict__ trans,    // (C+2,C+2)
    const int*   __restrict__ length,   // (B,)
    float*       __restrict__ total_out)// (B,)
{
    __shared__ float smET[CC * CC];   // exp(trans[:64,:64]), 16 KB
    __shared__ float smF [16 * CC];   // fwd state for 16 sequences, 4 KB
    __shared__ float smM [16];        // per-sequence row max
    __shared__ int   smLen[16];

    const int lane   = threadIdx.x;        // 0..31 (wave32)
    const int group  = blockIdx.x;         // 0..31
    const int batch0 = group * 16;
    const int half   = lane >> 4;          // 0 | 1
    const int l15    = lane & 15;

    // ---- precompute ET = exp(trans) into LDS -----------------------------
    for (int i = lane; i < CC * CC; i += 32) {
        int j = i >> 6, k = i & 63;
        smET[i] = __expf(trans[j * CP2 + k]);
    }
    if (lane < 16) smLen[lane] = length[batch0 + lane];
    __syncthreads();

    // ---- build constant B fragments (ET) once: Bfrag[kchunk][ntile] ------
    // B 32x16 f16, contiguous k order used here (A matches): lanes 0-15 hold K=0..15 (2 per VGPR), lanes 16-31 K=16..31.
    v16h Bfrag[2][4];
#pragma unroll
    for (int c = 0; c < 2; ++c)
#pragma unroll
        for (int n = 0; n < 4; ++n) {
            v16h tmp;
#pragma unroll
            for (int i = 0; i < 16; ++i) {
                int K = c * 32 + half * 16 + i;
                int N = n * 16 + l15;
                tmp[i] = (_Float16)smET[K * CC + N];
            }
            Bfrag[c][n] = tmp;
        }

    // ---- step 0: fwd0[k] = x_0[k] + trans[START,k] + log(C) --------------
    const float LOGC = 4.1588830833596715f; // log(64)
    for (int i = lane; i < 16 * CC; i += 32) {
        int m = i >> 6, k = i & 63;
        smF[i] = inputs[(size_t)(batch0 + m) * TT * CC + k]
               + trans[START_TAG * CP2 + k] + LOGC;
    }
    __syncthreads();

    // ---- software-pipelined emission loads: xc holds x_t -----------------
    float xc[32];
#pragma unroll
    for (int i = 0; i < 32; ++i) {
        int n = i >> 3, v = i & 7;
        int M = v + half * 8, N = n * 16 + l15;
        xc[i] = inputs[((size_t)(batch0 + M) * TT + 1) * CC + N];
    }

    for (int t = 1; t < TT; ++t) {
        // prefetch x_{t+1} early; latency hidden under exp/wmma/log below
        float xn[32];
#pragma unroll
        for (int i = 0; i < 32; ++i) {
            int n = i >> 3, v = i & 7;
            int M = v + half * 8, N = n * 16 + l15;
            xn[i] = (t + 1 < TT)
                  ? inputs[((size_t)(batch0 + M) * TT + (t + 1)) * CC + N]
                  : 0.0f;
        }

        // per-sequence max of fwd row l15 (halves combined via lane shuffle)
        float pm = -3.0e38f;
#pragma unroll
        for (int k = 0; k < 32; ++k)
            pm = fmaxf(pm, smF[l15 * CC + half * 32 + k]);
        float mrow = fmaxf(pm, __shfl_xor(pm, 16, 32));
        if (half == 0) smM[l15] = mrow;
        __syncthreads();

        // A fragments: p = exp(fwd - m) in f16, WMMA-accumulate over K
        v8f acc[4] = {};
#pragma unroll
        for (int c = 0; c < 2; ++c) {
            v16h Af;
            const float mb = smM[l15];
#pragma unroll
            for (int i = 0; i < 16; ++i) {
                // 16-bit A 16x32 layout (matches B): element i -> K = half*16 + i
                int K = c * 32 + half * 16 + i;
                Af[i] = (_Float16)__expf(smF[l15 * CC + K] - mb);
            }
#pragma unroll
            for (int n = 0; n < 4; ++n)
                acc[n] = __builtin_amdgcn_wmma_f32_16x16x32_f16(
                    false, Af, false, Bfrag[c][n], (short)0, acc[n], false, false);
        }
        __syncthreads();

        // write back: new = m + x_t + log(acc), frozen once t >= len[b]
#pragma unroll
        for (int n = 0; n < 4; ++n)
#pragma unroll
            for (int v = 0; v < 8; ++v) {
                int M = v + half * 8;
                int N = n * 16 + l15;
                float newv = smM[M] + xc[n * 8 + v] + __logf(acc[n][v]);
                if (t < smLen[M]) smF[M * CC + N] = newv;
            }
        __syncthreads();

#pragma unroll
        for (int i = 0; i < 32; ++i) xc[i] = xn[i];
    }

    // ---- termination: total = clip + log(sum exp(fwd + trans[:,END] - clip))
    {
        float pm = -3.0e38f;
#pragma unroll
        for (int k = 0; k < 32; ++k) {
            int kk = half * 32 + k;
            pm = fmaxf(pm, smF[l15 * CC + kk] + trans[kk * CP2 + END_TAG]);
        }
        float clip = truncf(fmaxf(pm, __shfl_xor(pm, 16, 32)));
        float ps = 0.0f;
#pragma unroll
        for (int k = 0; k < 32; ++k) {
            int kk = half * 32 + k;
            ps += __expf(smF[l15 * CC + kk] + trans[kk * CP2 + END_TAG] - clip);
        }
        float s = ps + __shfl_xor(ps, 16, 32);
        if (half == 0) total_out[batch0 + l15] = clip + __logf(s);
    }
}

// ---------------------------------------------------------------------------
// Real (gold-path) score: one block per sequence.
// ---------------------------------------------------------------------------
__global__ __launch_bounds__(256) void crf_real_kernel(
    const float* __restrict__ inputs, const float* __restrict__ trans,
    const int* __restrict__ tags, const int* __restrict__ length,
    float* __restrict__ real_out)
{
    const int b   = blockIdx.x;
    const int len = length[b];
    __shared__ float red[256];

    float s = 0.0f;
    for (int t = threadIdx.x; t < len; t += 256) {
        int tag  = tags[b * TT + t];
        int prev = (t == 0) ? START_TAG : tags[b * TT + t - 1];
        s += inputs[(size_t)b * TT * CC + (size_t)t * CC + tag]
           + trans[prev * CP2 + tag];
    }
    red[threadIdx.x] = s;
    __syncthreads();
    for (int off = 128; off > 0; off >>= 1) {
        if (threadIdx.x < off) red[threadIdx.x] += red[threadIdx.x + off];
        __syncthreads();
    }
    if (threadIdx.x == 0) {
        int last = tags[b * TT + len - 1];
        real_out[b] = red[0] + trans[last * CP2 + END_TAG];
    }
}

// ---------------------------------------------------------------------------
// Final reduction: sum(total - real) / sum(length)  -> single float
// ---------------------------------------------------------------------------
__global__ __launch_bounds__(512) void crf_reduce_kernel(
    const float* __restrict__ total, const float* __restrict__ real,
    const int* __restrict__ length, float* __restrict__ out)
{
    __shared__ float rs[512];
    __shared__ int   rl[512];
    int i = threadIdx.x;
    rs[i] = total[i] - real[i];
    rl[i] = length[i];
    __syncthreads();
    for (int off = 256; off > 0; off >>= 1) {
        if (i < off) { rs[i] += rs[i + off]; rl[i] += rl[i + off]; }
        __syncthreads();
    }
    if (i == 0) out[0] = rs[0] / (float)rl[0];
}

// ---------------------------------------------------------------------------
extern "C" void kernel_launch(void* const* d_in, const int* in_sizes, int n_in,
                              void* d_out, int out_size, void* d_ws, size_t ws_size,
                              hipStream_t stream) {
    const float* inputs = (const float*)d_in[0];   // (512,1024,64) f32
    const float* trans  = (const float*)d_in[1];   // (66,66) f32
    const int*   tags   = (const int*)  d_in[2];   // (512,1024) i32
    const int*   length = (const int*)  d_in[3];   // (512,) i32
    float*       out    = (float*)d_out;

    float* total = (float*)d_ws;          // 512 floats
    float* real  = total + BB;            // 512 floats

    crf_forward_kernel<<<BB / 16, 32, 0, stream>>>(inputs, trans, length, total);
    crf_real_kernel  <<<BB, 256, 0, stream>>>(inputs, trans, tags, length, real);
    crf_reduce_kernel<<<1, BB, 0, stream>>>(total, real, length, out);
}
